// RiemannianIFNode_38852274159632
// MI455X (gfx1250) — matrix operand order repaired and checked by reading, not compiled, not run on hardware
//
#include <hip/hip_runtime.h>
#include <hip/hip_bf16.h>
#include <math.h>

typedef __attribute__((ext_vector_type(4)))  float    v4f;
typedef __attribute__((ext_vector_type(8)))  float    v8f;
typedef __attribute__((ext_vector_type(16))) _Float16 v16h;

#define D_DIM 256
#define V_TH  1.0f
#define EPS_F 1e-8f

// ---------------------------------------------------------------------------
// Kernel 1: integrate-and-fire scan over T timesteps.
// Pure streaming: preload all T float4's (independent loads -> max MLP),
// then sequential accumulate/threshold, NT stores (single-use 1GB streams).
// ---------------------------------------------------------------------------
template <int T>
__global__ __launch_bounds__(256) void if_scan_kernel(const v4f* __restrict__ x,
                                                      v4f* __restrict__ s,
                                                      int total4) {
    int i = blockIdx.x * blockDim.x + threadIdx.x;
    if (i >= total4) return;

    v4f xs[T];
#pragma unroll
    for (int t = 0; t < T; ++t)
        xs[t] = __builtin_nontemporal_load(x + (size_t)t * total4 + i);

    v4f mem = {0.f, 0.f, 0.f, 0.f};
#pragma unroll
    for (int t = 0; t < T; ++t) {
        mem += xs[t];
        v4f sp;
#pragma unroll
        for (int j = 0; j < 4; ++j)
            sp[j] = mem[j] > V_TH ? 1.0f : 0.0f;
        mem -= sp;
        __builtin_nontemporal_store(sp, s + (size_t)t * total4 + i);
    }
}

// Fallback for T != 16 (keeps kernel_launch deterministic for any T).
__global__ __launch_bounds__(256) void if_scan_dyn_kernel(const v4f* __restrict__ x,
                                                          v4f* __restrict__ s,
                                                          int total4, int T) {
    int i = blockIdx.x * blockDim.x + threadIdx.x;
    if (i >= total4) return;
    v4f mem = {0.f, 0.f, 0.f, 0.f};
    for (int t = 0; t < T; ++t) {
        v4f xt = __builtin_nontemporal_load(x + (size_t)t * total4 + i);
        mem += xt;
        v4f sp;
#pragma unroll
        for (int j = 0; j < 4; ++j)
            sp[j] = mem[j] > V_TH ? 1.0f : 0.0f;
        mem -= sp;
        __builtin_nontemporal_store(sp, s + (size_t)t * total4 + i);
    }
}

// ---------------------------------------------------------------------------
// Kernel 2: Lorentz expmap. One block = 16 rows of D=256.
// Wave 0 computes the Minkowski inner products <v,v>_L of all 16 rows as the
// diagonal of a 16x16 Gram matrix via 8x v_wmma_f32_16x16x32_f16 (A = v with
// k=0 negated, B = v). Then all 8 waves do the fp32 elementwise expmap.
//
// A layout (16-bit A 16x32, per ISA 7.12.2): lane L (m=L%16, h=L/16) holds
//   halves 0..7  -> K = s*32 + h*8 + j
//   halves 8..15 -> K = s*32 + 16 + h*8 + j
// B layout (dense 32x16): same K order as A; lane L
// (n=L%16, h=L/16) holds halves 0..7 -> K = s*32 + h*8 + j, 8..15 -> K = s*32 + 16 + h*8 + j.
// C/D diag: m<8 -> lane m, vgpr m ; m>=8 -> lane m+16, vgpr m-8.
// ---------------------------------------------------------------------------
__global__ __launch_bounds__(256) void lorentz_expmap_wmma_kernel(const float* __restrict__ v,
                                                                  const float* __restrict__ z,
                                                                  float* __restrict__ zout,
                                                                  int N) {
    __shared__ float sIp[16];
    const int rowBase = blockIdx.x * 16;

    if (threadIdx.x < 32) {            // wave 0 only; EXEC all-1s inside
        const int L = threadIdx.x;
        const int m = L & 15;
        const int h = L >> 4;
        int row = rowBase + m;
        if (row >= N) row = N - 1;     // clamp (pad rows harmlessly recomputed)
        const float* vr = v + (size_t)row * D_DIM;

        v8f c = {0.f, 0.f, 0.f, 0.f, 0.f, 0.f, 0.f, 0.f};
#pragma unroll
        for (int s8 = 0; s8 < 8; ++s8) {
            const int kb = s8 * 32;
            float af[16], bf[16];
#pragma unroll
            for (int j = 0; j < 8; ++j) {
                af[j]     = vr[kb + h * 8 + j];
                af[j + 8] = vr[kb + 16 + h * 8 + j];
            }
#pragma unroll
            for (int j = 0; j < 8; ++j) {
                bf[j]     = vr[kb + h * 8 + j];
                bf[j + 8] = vr[kb + 16 + h * 8 + j];
            }
            // Minkowski metric: negate the k==0 term (only s8==0, lanes 0-15,
            // half 0 of A holds K=0).
            if (s8 == 0 && h == 0) af[0] = -af[0];

            v16h a, b;
#pragma unroll
            for (int j = 0; j < 16; ++j) {
                a[j] = (_Float16)af[j];
                b[j] = (_Float16)bf[j];
            }
            c = __builtin_amdgcn_wmma_f32_16x16x32_f16(
                    /*neg_a=*/false, a, /*neg_b=*/false, b,
                    /*c_mod=*/(short)0, c, /*reuse_a=*/false, /*reuse_b=*/false);
        }

        // Extract Gram diagonal.
        float carr[8];
#pragma unroll
        for (int j = 0; j < 8; ++j) carr[j] = c[j];
        const int sel = L & 7;
        float ipv = carr[0];
#pragma unroll
        for (int j = 1; j < 8; ++j)
            if (sel == j) ipv = carr[j];
        if (L < 8)        sIp[L]      = ipv;   // m = L
        else if (L >= 24) sIp[L - 16] = ipv;   // m = L - 16
    }
    __syncthreads();

    // Elementwise expmap: 16 rows x 64 float4 = 1024 float4, 4 per thread.
#pragma unroll
    for (int j = 0; j < 4; ++j) {
        const int lin = threadIdx.x + j * 256;
        const int r   = lin >> 6;   // 0..15
        const int c4  = lin & 63;   // 0..63
        const int row = rowBase + r;
        if (row < N) {
            const float ip = sIp[r];
            const float vn = sqrtf(fmaxf(ip, EPS_F));
            const float e  = expf(vn);
            const float ei = 1.0f / e;
            const float ch  = 0.5f * (e + ei);
            const float shv = 0.5f * (e - ei) / vn;
            const v4f vv = *((const v4f*)(v + (size_t)row * D_DIM) + c4);
            const v4f zz = *((const v4f*)(z + (size_t)row * D_DIM) + c4);
            v4f o;
#pragma unroll
            for (int k = 0; k < 4; ++k)
                o[k] = ch * zz[k] + shv * vv[k];
            __builtin_nontemporal_store(o, (v4f*)(zout + (size_t)row * D_DIM) + c4);
        }
    }
}

// ---------------------------------------------------------------------------
// Host launcher
// ---------------------------------------------------------------------------
extern "C" void kernel_launch(void* const* d_in, const int* in_sizes, int n_in,
                              void* d_out, int out_size, void* d_ws, size_t ws_size,
                              hipStream_t stream) {
    (void)n_in; (void)out_size; (void)d_ws; (void)ws_size;

    const float* x = (const float*)d_in[0];   // [T, N, D]
    const float* v = (const float*)d_in[1];   // [N, D]
    const float* z = (const float*)d_in[2];   // [N, D]

    const long long ND = in_sizes[1];               // N * D
    const int T = (int)(in_sizes[0] / (ND ? ND : 1));
    const int N = (int)(ND / D_DIM);

    float* s_out = (float*)d_out;                              // [T, N, D]
    float* z_out = (float*)d_out + (size_t)T * (size_t)ND;     // [N, D]

    // 1) IF scan (streaming, NT hints).
    const int total4 = (int)(ND / 4);
    dim3 blk(256);
    dim3 grd((total4 + 255) / 256);
    if (T == 16) {
        if_scan_kernel<16><<<grd, blk, 0, stream>>>((const v4f*)x, (v4f*)s_out, total4);
    } else {
        if_scan_dyn_kernel<<<grd, blk, 0, stream>>>((const v4f*)x, (v4f*)s_out, total4, T);
    }

    // 2) Lorentz expmap (WMMA Gram-diagonal reduction + fp32 elementwise).
    dim3 grd2((N + 15) / 16);
    lorentz_expmap_wmma_kernel<<<grd2, blk, 0, stream>>>(v, z, z_out, N);
}
